// MultiHeadAttention_850403524913
// MI455X (gfx1250) — matrix operand order repaired and checked by reading, not compiled, not run on hardware
//
#include <hip/hip_runtime.h>
#include <hip/hip_bf16.h>

typedef __attribute__((ext_vector_type(16))) _Float16 v16h;
typedef __attribute__((ext_vector_type(8)))  _Float16 v8h;
typedef __attribute__((ext_vector_type(8)))  float    v8f;

namespace cfg {
constexpr int B  = 2;
constexpr int N  = 2048;
constexpr int C  = 768;
constexpr int H  = 12;
constexpr int DH = 64;
constexpr int BN = B * N;      // 4096 rows
}

__device__ __forceinline__ v8f wmma_f32_16x16x32(v16h a, v16h b, v8f c) {
  return __builtin_amdgcn_wmma_f32_16x16x32_f16(
      /*neg_a=*/false, a, /*neg_b=*/false, b,
      /*c_mod=*/(short)0, c, /*reuse_a=*/false, /*reuse_b=*/false);
}

// A-matrix fragment: 16(M) x 32(K) f16, source row-major with K contiguous.
// ISA layout: lanes 0-15 -> M=lane, K = 0-7 then 16-23; lanes 16-31 -> K = 8-15 then 24-31.
__device__ __forceinline__ v16h load_frag_a(const _Float16* base, int ld) {
  const int lane = threadIdx.x & 31;
  const int row  = lane & 15;
  const int kb   = (lane >> 4) << 3;               // 0 or 8
  const _Float16* p = base + row * ld + kb;
  v8h lo = *(const v8h*)(p);
  v8h hi = *(const v8h*)(p + 16);
  v16h r;
#pragma unroll
  for (int i = 0; i < 8; ++i) { r[i] = lo[i]; r[i + 8] = hi[i]; }
  return r;
}

// B-matrix fragment: 32(K) x 16(N) f16, sourced from B^T stored as N x K (K contiguous).
// ISA layout: lanes 0-15 -> N=lane, K = 0-7 then 16-23; lanes 16-31 -> K = 8-15 then 24-31.
__device__ __forceinline__ v16h load_frag_b(const _Float16* base, int ld) {
  const int lane = threadIdx.x & 31;
  const int col  = lane & 15;
  const int kb   = (lane >> 4) << 3;               // 0 or 8
  const _Float16* p = base + col * ld + kb;
  v8h lo = *(const v8h*)(p);
  v8h hi = *(const v8h*)(p + 16);
  v16h r;
#pragma unroll
  for (int i = 0; i < 8; ++i) { r[i] = lo[i]; r[i + 8] = hi[i]; }
  return r;
}

// ---- CDNA5 async memory->LDS DMA (ASYNCcnt tracked, per-lane addresses) ----
__device__ __forceinline__ void async_copy_b128(const void* gsrc, void* lds_dst) {
  unsigned l = (unsigned)(unsigned long long)lds_dst;   // low 32 bits == LDS offset
  asm volatile("global_load_async_to_lds_b128 %0, %1, off"
               :: "v"(l), "v"(gsrc)
               : "memory");
}
__device__ __forceinline__ void wait_async0() { asm volatile("s_wait_asynccnt 0x0" ::: "memory"); }
__device__ __forceinline__ void wait_async2() { asm volatile("s_wait_asynccnt 0x2" ::: "memory"); }
__device__ __forceinline__ void wait_async3() { asm volatile("s_wait_asynccnt 0x3" ::: "memory"); }

// ---------------- prep kernels ----------------
__global__ __launch_bounds__(256) void cvt_f32_f16(const float* __restrict__ in,
                                                   _Float16* __restrict__ out, int n) {
  int i = blockIdx.x * 256 + threadIdx.x;
  if (i < n) out[i] = (_Float16)in[i];
}

// in: W[k][n] (768x768)  -> out: Wt[n][k] f16
__global__ __launch_bounds__(256) void transpose_w(const float* __restrict__ in,
                                                   _Float16* __restrict__ out) {
  int i = blockIdx.x * 256 + threadIdx.x;
  if (i < cfg::C * cfg::C) {
    int n = i / cfg::C, k = i % cfg::C;
    out[i] = (_Float16)in[k * cfg::C + n];
  }
}

// ---------------- QKV projection ----------------
// grid: (BN/128, H)  block: 256 (8 waves). BM=128, BN=64 (one head), BK=32.
// One fully-specialized pass per matrix (Q,K,V): 4 live accumulator tuples per wave.
__global__ __launch_bounds__(256) void qkv_gemm(
    const _Float16* __restrict__ x16,                 // [4096, 768]
    const _Float16* __restrict__ wqt,                 // [768(out), 768(in)]
    const _Float16* __restrict__ wkt,
    const _Float16* __restrict__ wvt,
    const float* __restrict__ bq, const float* __restrict__ bk, const float* __restrict__ bv,
    _Float16* __restrict__ qh,                        // [B,H,N,DH] (pre-scaled 1/8)
    _Float16* __restrict__ kh,                        // [B,H,N,DH]
    _Float16* __restrict__ vth) {                     // [B,H,DH,N] (transposed)
  __shared__ __align__(64) _Float16 Ab[2 * 128 * 32];      // 16 KB, buffer stride 4096 elems
  __shared__ __align__(64) _Float16 Bb[2 * 64 * 32];       //  8 KB, buffer stride 2048 elems
  constexpr unsigned AS = 128 * 32;
  constexpr unsigned BS = 64 * 32;

  const int tid  = threadIdx.x;
  const int w    = tid >> 5;
  const int lane = tid & 31;
  const int m_blk = blockIdx.x * 128;
  const int h     = blockIdx.y;
  const int n_blk = h * 64;

  const int mi0 = (w >> 1) * 2;                      // 0,2,4,6
  const int ni0 = (w & 1) * 2;                       // 0 or 2

  const int arow = tid >> 2;                         // 0..63
  const int acol = (tid & 3) * 8;                    // 0,8,16,24
  const unsigned ldst = (unsigned)(arow * 32 + acol);

  const int nl = lane & 15;
  const int m0 = (lane >> 4) * 8;

  // 128-row block never crosses the batch boundary (2048 % 128 == 0)
  const int  bb  = m_blk >> 11;
  const long hb  = (long)(bb * cfg::H + h);
  const int  nsb = m_blk & (cfg::N - 1);

#pragma unroll
  for (int mtx = 0; mtx < 3; ++mtx) {                // fully unrolled: mtx is constexpr
    const _Float16* wt = (mtx == 0) ? wqt : (mtx == 1) ? wkt : wvt;
    v8f acc00 = {}, acc01 = {}, acc10 = {}, acc11 = {};

    // running global source pointers (advance by BK=32 per step)
    const _Float16* xg0 = x16 + (m_blk + arow) * cfg::C + acol;
    const _Float16* xg1 = xg0 + 64 * cfg::C;
    const _Float16* wg  = wt + (n_blk + arow) * cfg::C + acol;

    auto issue = [&](unsigned ao, unsigned bo) {     // 3 async b128 per thread
      async_copy_b128(xg0, Ab + ao + ldst);
      async_copy_b128(xg1, Ab + ao + 64 * 32 + ldst);
      async_copy_b128(wg,  Bb + bo + ldst);
      xg0 += 32; xg1 += 32; wg += 32;
    };
    auto compute = [&](unsigned ao, unsigned bo) {
      v16h af0 = load_frag_a(Ab + ao + (mi0 + 0) * 16 * 32, 32);
      v16h af1 = load_frag_a(Ab + ao + (mi0 + 1) * 16 * 32, 32);
      v16h bf0 = load_frag_b(Bb + bo + (ni0 + 0) * 16 * 32, 32);
      v16h bf1 = load_frag_b(Bb + bo + (ni0 + 1) * 16 * 32, 32);
      acc00 = wmma_f32_16x16x32(af0, bf0, acc00);
      acc01 = wmma_f32_16x16x32(af1, bf0, acc01);
      acc10 = wmma_f32_16x16x32(af0, bf1, acc10);
      acc11 = wmma_f32_16x16x32(af1, bf1, acc11);
    };

    unsigned ao = 0, bo = 0;
    issue(ao, bo);                                   // fill buffer 0
#pragma unroll 1
    for (int kt = 0; kt < cfg::C / 32 - 1; ++kt) {   // single-body steady state
      issue(ao ^ AS, bo ^ BS);                       // fill other buffer
      wait_async3();                                 // current 3 done, next 3 in flight
      __syncthreads();
      compute(ao, bo);
      __syncthreads();
      ao ^= AS; bo ^= BS;
    }
    wait_async0();
    __syncthreads();
    compute(ao, bo);

    // branch-free specialized epilogue (mtx is a compile-time constant here)
#pragma unroll
    for (int j = 0; j < 2; ++j) {
      const int d = (ni0 + j) * 16 + nl;             // 0..63 within the head
      const float bv_ = ((mtx == 0) ? bq : (mtx == 1) ? bk : bv)[n_blk + d];
#pragma unroll
      for (int im = 0; im < 2; ++im) {
        const v8f a = (j == 0) ? ((im == 0) ? acc00 : acc01)
                               : ((im == 0) ? acc10 : acc11);
        const int ns0 = nsb + (mi0 + im) * 16 + m0;
        if (mtx == 2) {
          // transposed V: 8 contiguous f16 per lane -> one b128 store
          _Float16* dst = vth + (hb * cfg::DH + d) * cfg::N + ns0;
          v8h pk;
#pragma unroll
          for (int r = 0; r < 8; ++r) pk[r] = (_Float16)(a[r] + bv_);
          *(v8h*)dst = pk;
        } else {
          _Float16* dst = ((mtx == 0) ? qh : kh) + (hb * cfg::N + ns0) * cfg::DH + d;
          const float sc = (mtx == 0) ? 0.125f : 1.0f;
#pragma unroll
          for (int r = 0; r < 8; ++r)
            dst[r * cfg::DH] = (_Float16)((a[r] + bv_) * sc);
        }
      }
    }
    __syncthreads();                                 // keep passes in lockstep
  }
}

// ---------------- flash attention ----------------
// grid: (N/128, B*H)  block: 256 (8 waves); each wave owns a 16-row query tile.
// K/V kv-tiles staged once per block in LDS via async DMA (8x reuse across waves).
__global__ __launch_bounds__(256) void attn_fwd(
    const _Float16* __restrict__ qh,   // [B*H, N, 64] pre-scaled
    const _Float16* __restrict__ kh,   // [B*H, N, 64]
    const _Float16* __restrict__ vth,  // [B*H, 64, N]
    _Float16* __restrict__ y16) {      // [B*N, C]
  __shared__ __align__(64) _Float16 Kt[2 * 32 * 64];     // 8 KB (rows kv, d contiguous)
  __shared__ __align__(64) _Float16 Vt[2 * 64 * 32];     // 8 KB (rows d, kv contiguous)
  __shared__ __align__(64) _Float16 pscr[8][16 * 32];    // 8 KB wave-private P scratch
  constexpr unsigned TS = 32 * 64;                       // == 64*32, both tile strides

  const int tid  = threadIdx.x;
  const int w    = tid >> 5;
  const int lane = tid & 31;
  const int bh   = blockIdx.y;
  const int q0   = blockIdx.x * 128 + w * 16;

  const _Float16* qptr = qh  + ((long)bh * cfg::N + q0) * cfg::DH;
  const _Float16* kptr = kh  + (long)bh * cfg::N * cfg::DH;
  const _Float16* vptr = vth + (long)bh * cfg::DH * cfg::N;

  v16h aq0 = load_frag_a(qptr, cfg::DH);       // d = 0..31
  v16h aq1 = load_frag_a(qptr + 32, cfg::DH);  // d = 32..63

  v8f acc[4] = {};
  float rm[8], rl[8];
#pragma unroll
  for (int r = 0; r < 8; ++r) { rm[r] = -1e30f; rl[r] = 0.0f; }

  _Float16* ps = pscr[w];
  const int nl = lane & 15;
  const int m0 = (lane >> 4) * 8;

  const int krow = tid >> 3, kcol = (tid & 7) * 8;   // 32x64 tile, 1 b128/thread
  const int vrow = tid >> 2, vcol = (tid & 3) * 8;   // 64x32 tile, 1 b128/thread
  const unsigned kdst = (unsigned)(krow * 64 + kcol);
  const unsigned vdst = (unsigned)(vrow * 32 + vcol);

  // running global source pointers (advance by 32 kv per step)
  const _Float16* kg = kptr + (long)krow * cfg::DH + kcol;
  const _Float16* vg = vptr + (long)vrow * cfg::N + vcol;

  auto issue = [&](unsigned bo) {                    // 2 async b128 per thread
    async_copy_b128(kg, Kt + bo + kdst);
    async_copy_b128(vg, Vt + bo + vdst);
    kg += 32 * cfg::DH; vg += 32;
  };

  auto step = [&](unsigned bo) {
    // scores: two 16x16 tiles covering 32 keys, full DH=64 reduction
    v8f s0 = {}, s1 = {};
    s0 = wmma_f32_16x16x32(aq0, load_frag_b(Kt + bo, 64), s0);
    s0 = wmma_f32_16x16x32(aq1, load_frag_b(Kt + bo + 32, 64), s0);
    s1 = wmma_f32_16x16x32(aq0, load_frag_b(Kt + bo + 16 * 64, 64), s1);
    s1 = wmma_f32_16x16x32(aq1, load_frag_b(Kt + bo + 16 * 64 + 32, 64), s1);

    // online softmax; C-layout row = VGPR index, cols striped across 16 lanes
#pragma unroll
    for (int r = 0; r < 8; ++r) {
      float mx = fmaxf(s0[r], s1[r]);
      mx = fmaxf(mx, __shfl_xor(mx, 1));
      mx = fmaxf(mx, __shfl_xor(mx, 2));
      mx = fmaxf(mx, __shfl_xor(mx, 4));
      mx = fmaxf(mx, __shfl_xor(mx, 8));
      const float mnew = fmaxf(rm[r], mx);
      const float corr = __expf(rm[r] - mnew);
      const float p0 = __expf(s0[r] - mnew);
      const float p1 = __expf(s1[r] - mnew);
      float psum = p0 + p1;
      psum += __shfl_xor(psum, 1);
      psum += __shfl_xor(psum, 2);
      psum += __shfl_xor(psum, 4);
      psum += __shfl_xor(psum, 8);
      rl[r] = rl[r] * corr + psum;
      rm[r] = mnew;
      acc[0][r] *= corr; acc[1][r] *= corr; acc[2][r] *= corr; acc[3][r] *= corr;
      ps[(m0 + r) * 32 + nl]      = (_Float16)p0;
      ps[(m0 + r) * 32 + 16 + nl] = (_Float16)p1;
    }
    // same-wave DS ops are in-order (ISA 7.3); just stop compiler reordering
    __asm volatile("" ::: "memory");
    v16h pf = load_frag_a(ps, 32);               // P as 16x32 A-fragment
    __asm volatile("" ::: "memory");
#pragma unroll
    for (int c = 0; c < 4; ++c)
      acc[c] = wmma_f32_16x16x32(pf, load_frag_b(Vt + bo + (c * 16) * 32, 32), acc[c]);
  };

  constexpr int NSTEP = cfg::N / 32;               // 64
  unsigned bo = 0;
  issue(bo);
#pragma unroll 1
  for (int st = 0; st < NSTEP - 1; ++st) {         // single-body steady state
    issue(bo ^ TS);
    wait_async2();
    __syncthreads();
    step(bo);
    __syncthreads();
    bo ^= TS;
  }
  wait_async0();
  __syncthreads();
  step(bo);

  // epilogue: strength-reduced addressing
  float inv[8];
#pragma unroll
  for (int r = 0; r < 8; ++r) inv[r] = 1.0f / rl[r];
  const int b = bh / cfg::H, h = bh % cfg::H;
  _Float16* ybase = y16 + ((long)(b * cfg::N + q0 + m0)) * cfg::C + h * 64;
#pragma unroll
  for (int c = 0; c < 4; ++c) {
    const int d = c * 16 + nl;
#pragma unroll
    for (int r = 0; r < 8; ++r)
      ybase[(long)r * cfg::C + d] = (_Float16)(acc[c][r] * inv[r]);
  }
}

// ---------------- output projection ----------------
// grid: (BN/128, C/64)  block 256. BM=128, BN=64, 2x2 register blocking per wave.
__global__ __launch_bounds__(256) void proj_gemm(
    const _Float16* __restrict__ y16,   // [4096, 768]
    const _Float16* __restrict__ wpt,   // [768(out), 768(in)]
    const float* __restrict__ bp,
    float* __restrict__ out) {          // [4096, 768] f32
  __shared__ __align__(64) _Float16 Ab[2 * 128 * 32];
  __shared__ __align__(64) _Float16 Bb[2 * 64 * 32];
  constexpr unsigned AS = 128 * 32;
  constexpr unsigned BS = 64 * 32;

  const int tid  = threadIdx.x;
  const int w    = tid >> 5;
  const int lane = tid & 31;
  const int m_blk = blockIdx.x * 128;
  const int n_blk = blockIdx.y * 64;

  const int mi0 = (w >> 1) * 2;
  const int ni0 = (w & 1) * 2;
  v8f acc00 = {}, acc01 = {}, acc10 = {}, acc11 = {};

  const int arow = tid >> 2;
  const int acol = (tid & 3) * 8;
  const unsigned ldst = (unsigned)(arow * 32 + acol);

  const _Float16* yg0 = y16 + (m_blk + arow) * cfg::C + acol;
  const _Float16* yg1 = yg0 + 64 * cfg::C;
  const _Float16* wg  = wpt + (n_blk + arow) * cfg::C + acol;

  auto issue = [&](unsigned ao, unsigned bo) {
    async_copy_b128(yg0, Ab + ao + ldst);
    async_copy_b128(yg1, Ab + ao + 64 * 32 + ldst);
    async_copy_b128(wg,  Bb + bo + ldst);
    yg0 += 32; yg1 += 32; wg += 32;
  };
  auto compute = [&](unsigned ao, unsigned bo) {
    v16h af0 = load_frag_a(Ab + ao + (mi0 + 0) * 16 * 32, 32);
    v16h af1 = load_frag_a(Ab + ao + (mi0 + 1) * 16 * 32, 32);
    v16h bf0 = load_frag_b(Bb + bo + (ni0 + 0) * 16 * 32, 32);
    v16h bf1 = load_frag_b(Bb + bo + (ni0 + 1) * 16 * 32, 32);
    acc00 = wmma_f32_16x16x32(af0, bf0, acc00);
    acc01 = wmma_f32_16x16x32(af1, bf0, acc01);
    acc10 = wmma_f32_16x16x32(af0, bf1, acc10);
    acc11 = wmma_f32_16x16x32(af1, bf1, acc11);
  };

  unsigned ao = 0, bo = 0;
  issue(ao, bo);
#pragma unroll 1
  for (int kt = 0; kt < cfg::C / 32 - 1; ++kt) {
    issue(ao ^ AS, bo ^ BS);
    wait_async3();
    __syncthreads();
    compute(ao, bo);
    __syncthreads();
    ao ^= AS; bo ^= BS;
  }
  wait_async0();
  __syncthreads();
  compute(ao, bo);

  const int nl = lane & 15;
  const int m0 = (lane >> 4) * 8;
#pragma unroll
  for (int j = 0; j < 2; ++j) {
    const int n = n_blk + (ni0 + j) * 16 + nl;
    const float bpv = bp[n];
#pragma unroll
    for (int im = 0; im < 2; ++im) {
      const v8f a = (j == 0) ? ((im == 0) ? acc00 : acc01)
                             : ((im == 0) ? acc10 : acc11);
      float* dst = out + (long)(m_blk + (mi0 + im) * 16 + m0) * cfg::C + n;
#pragma unroll
      for (int r = 0; r < 8; ++r)
        dst[(long)r * cfg::C] = a[r] + bpv;
    }
  }
}

extern "C" void kernel_launch(void* const* d_in, const int* in_sizes, int n_in,
                              void* d_out, int out_size, void* d_ws, size_t ws_size,
                              hipStream_t stream) {
  (void)in_sizes; (void)n_in; (void)out_size; (void)ws_size;
  const float* x  = (const float*)d_in[0];
  const float* Wq = (const float*)d_in[1];
  const float* bq = (const float*)d_in[2];
  const float* Wk = (const float*)d_in[3];
  const float* bk = (const float*)d_in[4];
  const float* Wv = (const float*)d_in[5];
  const float* bv = (const float*)d_in[6];
  const float* Wp = (const float*)d_in[7];
  const float* bp = (const float*)d_in[8];
  float* out = (float*)d_out;

  char* ws = (char*)d_ws;
  _Float16* x16 = (_Float16*)ws; ws += (size_t)cfg::BN * cfg::C * 2;
  _Float16* wqt = (_Float16*)ws; ws += (size_t)cfg::C * cfg::C * 2;
  _Float16* wkt = (_Float16*)ws; ws += (size_t)cfg::C * cfg::C * 2;
  _Float16* wvt = (_Float16*)ws; ws += (size_t)cfg::C * cfg::C * 2;
  _Float16* wpt = (_Float16*)ws; ws += (size_t)cfg::C * cfg::C * 2;
  _Float16* qh  = (_Float16*)ws; ws += (size_t)cfg::BN * cfg::C * 2;
  _Float16* kh  = (_Float16*)ws; ws += (size_t)cfg::BN * cfg::C * 2;
  _Float16* vth = (_Float16*)ws; ws += (size_t)cfg::BN * cfg::C * 2;
  _Float16* y16 = (_Float16*)ws; ws += (size_t)cfg::BN * cfg::C * 2;

  const int nx = cfg::BN * cfg::C;            // 3,145,728
  cvt_f32_f16<<<(nx + 255) / 256, 256, 0, stream>>>(x, x16, nx);

  const int nw = cfg::C * cfg::C;             // 589,824
  transpose_w<<<(nw + 255) / 256, 256, 0, stream>>>(Wq, wqt);
  transpose_w<<<(nw + 255) / 256, 256, 0, stream>>>(Wk, wkt);
  transpose_w<<<(nw + 255) / 256, 256, 0, stream>>>(Wv, wvt);
  transpose_w<<<(nw + 255) / 256, 256, 0, stream>>>(Wp, wpt);

  qkv_gemm<<<dim3(cfg::BN / 128, cfg::H), 256, 0, stream>>>(
      x16, wqt, wkt, wvt, bq, bk, bv, qh, kh, vth);

  attn_fwd<<<dim3(cfg::N / 128, cfg::B * cfg::H), 256, 0, stream>>>(qh, kh, vth, y16);

  proj_gemm<<<dim3(cfg::BN / 128, cfg::C / 64), 256, 0, stream>>>(y16, wpt, bp, out);
}
